// Model_22814866276898
// MI455X (gfx1250) — matrix operand order repaired and checked by reading, not compiled, not run on hardware
//
#include <hip/hip_runtime.h>
#include <hip/hip_bf16.h>

// ---------------------------------------------------------------------------
// CDNA5 (gfx1250) capsule-network forward pass.
// All heavy math (convs, transposed convs, FC, capsule einsum) runs as
// implicit GEMM on v_wmma_f32_16x16x32_f16 with f32 accumulation; operand
// tiles are double-buffered global->LDS with global_load_async_to_lds_b128
// (ASYNCcnt) so the async DMA overlaps the WMMA pipeline.
// ---------------------------------------------------------------------------

typedef __attribute__((ext_vector_type(16))) _Float16 v16h;
typedef __attribute__((ext_vector_type(8)))  _Float16 v8h;
typedef __attribute__((ext_vector_type(8)))  float    v8f;

#define TILE_M 128
#define TILE_N 64
#define TILE_K 32
#define KPAD   8      // halves of padding per LDS row (row stride 80 B)

static __device__ __forceinline__ int imin(int a, int b) { return a < b ? a : b; }

// ---------------------------------------------------------------------------
// Core WMMA GEMM: C = A[M,Kp] x B[N,Kp]^T (both f16 row-major), f32 acc.
// 256 threads = 8 waves arranged 4(M) x 2(N); each wave computes a 32x32
// output tile as 2x2 WMMA accumulators -> 4 v_wmma per K-step.
// grid.z batches independent slices. Output addressing:
//   ldYn >  0 : Y[z*yZ + n*ldYn + m]                  (capsule einsum mode)
//   ldYn <= 0 : Y[z*yZ + ((n/HoWo)*M + m)*HoWo + pix] (NCHW conv mode)
// ---------------------------------------------------------------------------
__global__ __launch_bounds__(256)
void k_gemm_wmma(const _Float16* __restrict__ A, long aZ,
                 const _Float16* __restrict__ Bm, long bZ,
                 float* __restrict__ Y, long yZ,
                 int M, int N, int Kp,
                 int HoWo, int ldYn,
                 const float* __restrict__ bias, int relu)
{
    __shared__ __align__(16) _Float16 As[2][TILE_M][TILE_K + KPAD];
    __shared__ __align__(16) _Float16 Bs[2][TILE_N][TILE_K + KPAD];

    const int tid  = threadIdx.x;
    const int lane = tid & 31;
    const int wv   = tid >> 5;            // 0..7
    const int wm   = (wv & 3) * 32;       // wave M offset within block tile
    const int wn   = (wv >> 2) * 32;      // wave N offset within block tile

    const int m0 = blockIdx.y * TILE_M;
    const int n0 = blockIdx.x * TILE_N;
    const int z  = blockIdx.z;
    const _Float16* Az = A  + (long)z * aZ;
    const _Float16* Bz = Bm + (long)z * bZ;

    v8f acc00 = {}, acc01 = {}, acc10 = {}, acc11 = {};

    // cooperative tile-load addressing: A = 512 x 16B chunks (2/thread),
    // B = 256 x 16B chunks (1/thread)
    const int arow = tid >> 2;            // 0..63  (A rows arow and arow+64)
    const int seg  = (tid & 3) * 8;       // half-element offset within row
    const long aRow0 = (long)imin(m0 + arow,      M - 1) * Kp;
    const long aRow1 = (long)imin(m0 + arow + 64, M - 1) * Kp;
    const long bRow  = (long)imin(n0 + arow,      N - 1) * Kp;

    const int nk = Kp / TILE_K;

    // issue one K-tile of async global->LDS copies into buffer `buf`
    auto issue = [&](int k0, int buf) {
        unsigned la0 = (unsigned)(size_t)&As[buf][arow][seg];
        unsigned la1 = (unsigned)(size_t)&As[buf][arow + 64][seg];
        unsigned lb  = (unsigned)(size_t)&Bs[buf][arow][seg];
        unsigned ga0 = (unsigned)((aRow0 + k0 + seg) * 2);
        unsigned ga1 = (unsigned)((aRow1 + k0 + seg) * 2);
        unsigned gb  = (unsigned)((bRow  + k0 + seg) * 2);
        asm volatile("global_load_async_to_lds_b128 %0, %1, %2"
                     :: "v"(la0), "v"(ga0), "s"(Az) : "memory");
        asm volatile("global_load_async_to_lds_b128 %0, %1, %2"
                     :: "v"(la1), "v"(ga1), "s"(Az) : "memory");
        asm volatile("global_load_async_to_lds_b128 %0, %1, %2"
                     :: "v"(lb),  "v"(gb),  "s"(Bz) : "memory");
    };

    issue(0, 0);
    asm volatile("s_wait_asynccnt 0" ::: "memory");
    __syncthreads();

    for (int i = 0; i < nk; ++i) {
        const int cb = i & 1;
        if (i + 1 < nk) {
            issue((i + 1) * TILE_K, cb ^ 1);
            __builtin_prefetch(Az + aRow0 + (long)(i + 1) * TILE_K + seg, 0, 1);
            __builtin_prefetch(Bz + bRow  + (long)(i + 1) * TILE_K + seg, 0, 1);
        }

        // --- fragments (documented 16-bit A/B VGPR layouts) ---
        const int ar = lane & 15;
        const int ak = (lane >> 4) * 8;
        const int bk = (lane >> 4) * 8;
        v16h a0, a1, b0, b1;
        {
            v8h lo = *(const v8h*)&As[cb][wm + ar][ak];
            v8h hi = *(const v8h*)&As[cb][wm + ar][ak + 16];
#pragma unroll
            for (int t = 0; t < 8; ++t) { a0[t] = lo[t]; a0[t + 8] = hi[t]; }
        }
        {
            v8h lo = *(const v8h*)&As[cb][wm + 16 + ar][ak];
            v8h hi = *(const v8h*)&As[cb][wm + 16 + ar][ak + 16];
#pragma unroll
            for (int t = 0; t < 8; ++t) { a1[t] = lo[t]; a1[t + 8] = hi[t]; }
        }
        {
            v8h lo = *(const v8h*)&Bs[cb][wn + ar][bk];
            v8h hi = *(const v8h*)&Bs[cb][wn + ar][bk + 16];
#pragma unroll
            for (int t = 0; t < 8; ++t) { b0[t] = lo[t]; b0[t + 8] = hi[t]; }
        }
        {
            v8h lo = *(const v8h*)&Bs[cb][wn + 16 + ar][bk];
            v8h hi = *(const v8h*)&Bs[cb][wn + 16 + ar][bk + 16];
#pragma unroll
            for (int t = 0; t < 8; ++t) { b1[t] = lo[t]; b1[t + 8] = hi[t]; }
        }

        acc00 = __builtin_amdgcn_wmma_f32_16x16x32_f16(false, a0, false, b0,
                                                       (short)0, acc00, false, false);
        acc01 = __builtin_amdgcn_wmma_f32_16x16x32_f16(false, a0, false, b1,
                                                       (short)0, acc01, false, false);
        acc10 = __builtin_amdgcn_wmma_f32_16x16x32_f16(false, a1, false, b0,
                                                       (short)0, acc10, false, false);
        acc11 = __builtin_amdgcn_wmma_f32_16x16x32_f16(false, a1, false, b1,
                                                       (short)0, acc11, false, false);

        asm volatile("s_wait_asynccnt 0" ::: "memory");
        __syncthreads();
    }

    // --- store four 16x16 f32 tiles (C/D layout: n=lane&15, m=8*(lane>>4)+r) ---
    const int nloc = lane & 15;
    const int mloc = (lane >> 4) * 8;
    const long outBase = (long)z * yZ;
#pragma unroll
    for (int j = 0; j < 2; ++j) {
        const int n = n0 + wn + 16 * j + nloc;
        if (n >= N) continue;
        int bidx = 0, pix = 0;
        if (ldYn <= 0) { bidx = n / HoWo; pix = n - bidx * HoWo; }
#pragma unroll
        for (int i2 = 0; i2 < 2; ++i2) {
            const v8f& acc = (i2 == 0) ? (j == 0 ? acc00 : acc01)
                                       : (j == 0 ? acc10 : acc11);
#pragma unroll
            for (int r = 0; r < 8; ++r) {
                int m = m0 + wm + 16 * i2 + mloc + r;
                if (m < M) {
                    float v = acc[r];
                    if (bias) v += bias[m];
                    if (relu) v = v > 0.f ? v : 0.f;
                    long idx = (ldYn > 0)
                                 ? (outBase + (long)n * ldYn + m)
                                 : (outBase + ((long)bidx * M + m) * HoWo + pix);
                    Y[idx] = v;
                }
            }
        }
    }
}

// ---------------------------------------------------------------------------
// Operand packing / im2col kernels (f32 -> f16, K zero-padded to Kp)
// ---------------------------------------------------------------------------
__global__ void k_pack_w(const float* __restrict__ w, _Float16* __restrict__ A,
                         int M, int K, int Kp)
{
    long i = (long)blockIdx.x * blockDim.x + threadIdx.x;
    long total = (long)M * Kp;
    if (i >= total) return;
    int k = (int)(i % Kp);
    long m = i / Kp;
    A[i] = (k < K) ? (_Float16)w[m * K + k] : (_Float16)0.f;
}

// transposed-conv weights: w (Cin, Cout, 3, 3) -> A[o][c*9 + r*3 + s] with flip
__global__ void k_pack_wt(const float* __restrict__ w, _Float16* __restrict__ A,
                          int Cin, int Cout, int Kp)
{
    int K = Cin * 9;
    long i = (long)blockIdx.x * blockDim.x + threadIdx.x;
    long total = (long)Cout * Kp;
    if (i >= total) return;
    int k = (int)(i % Kp);
    int o = (int)(i / Kp);
    _Float16 v = (_Float16)0.f;
    if (k < K) {
        int c = k / 9, rr = (k % 9) / 3, ss = k % 3;
        v = (_Float16)w[((long)(c * Cout + o) * 3 + (2 - rr)) * 3 + (2 - ss)];
    }
    A[i] = v;
}

// im2col for 3x3 conv, pad 1: col[(b,ho,wo)][c*9+r*3+s]
__global__ void k_im2col(const float* __restrict__ x, _Float16* __restrict__ col,
                         int Bn, int C, int H, int W, int Ho, int Wo,
                         int stride, int K, int Kp)
{
    long i = (long)blockIdx.x * blockDim.x + threadIdx.x;
    long total = (long)Bn * Ho * Wo * Kp;
    if (i >= total) return;
    int k = (int)(i % Kp);
    long n = i / Kp;
    _Float16 v = (_Float16)0.f;
    if (k < K) {
        int wo = (int)(n % Wo); long t = n / Wo;
        int ho = (int)(t % Ho); int b = (int)(t / Ho);
        int c = k / 9, rr = (k % 9) / 3, ss = k % 3;
        int hi = ho * stride + rr - 1;
        int wi = wo * stride + ss - 1;
        if (hi >= 0 && hi < H && wi >= 0 && wi < W)
            v = (_Float16)x[(((long)b * C + c) * H + hi) * W + wi];
    }
    col[i] = v;
}

// im2col for transposed 3x3 conv (lhs_dilation=stride, pad k-1-1=1),
// with optional leading zero-pad row/col ("prepad") folded in.
__global__ void k_im2col_t(const float* __restrict__ x, _Float16* __restrict__ col,
                           int Bn, int C, int Hp, int Wp, int Ho, int Wo,
                           int stride, int prepad, int K, int Kp)
{
    long i = (long)blockIdx.x * blockDim.x + threadIdx.x;
    long total = (long)Bn * Ho * Wo * Kp;
    if (i >= total) return;
    int k = (int)(i % Kp);
    long n = i / Kp;
    _Float16 v = (_Float16)0.f;
    if (k < K) {
        int wo = (int)(n % Wo); long t = n / Wo;
        int ho = (int)(t % Ho); int b = (int)(t / Ho);
        int c = k / 9, rr = (k % 9) / 3, ss = k % 3;
        int hd = ho + rr - 1;
        int wd = wo + ss - 1;
        if (hd >= 0 && wd >= 0 && (hd % stride) == 0 && (wd % stride) == 0) {
            int h = hd / stride - prepad;
            int w = wd / stride - prepad;
            if (h >= 0 && h < Hp && w >= 0 && w < Wp)
                v = (_Float16)x[(((long)b * C + c) * Hp + h) * Wp + w];
        }
    }
    col[i] = v;
}

// im2col for the 3D caps conv: x (128,256,2,2) viewed as (B,1,D=256,2,2),
// kernel (3,3,3), strides (8,1,1), pads (0,1,1). n = b*128 + dd*4 + h*2 + w.
__global__ void k_im2col3d(const float* __restrict__ x, _Float16* __restrict__ col)
{
    long i = (long)blockIdx.x * blockDim.x + threadIdx.x;
    long total = 128L * 32 * 4 * 32;
    if (i >= total) return;
    int k = (int)(i & 31);
    long n = i >> 5;
    int hw = (int)(n & 3); int wo = hw & 1, ho = hw >> 1;
    int dd = (int)((n >> 2) & 31);
    int b  = (int)(n >> 7);
    _Float16 v = (_Float16)0.f;
    if (k < 27) {
        int kd = k / 9, rr = (k % 9) / 3, ss = k % 3;
        int di = dd * 8 + kd;
        int hi = ho + rr - 1, wi = wo + ss - 1;
        if (di < 256 && hi >= 0 && hi < 2 && wi >= 0 && wi < 2)
            v = (_Float16)x[(((long)b * 256 + di) * 2 + hi) * 2 + wi];
    }
    col[i] = v;
}

// feats (128,640,8) -> f16 B-matrices [r=640][b=128][Kp=32]
__global__ void k_pack_feats(const float* __restrict__ y4,  // (128,32,8,2,2)
                             const float* __restrict__ x3,  // (128,32,8,4,4)
                             _Float16* __restrict__ dst)
{
    long i = (long)blockIdx.x * blockDim.x + threadIdx.x;
    long total = 640L * 128 * 32;
    if (i >= total) return;
    int k = (int)(i % 32); long t = i / 32;
    int b = (int)(t % 128); int r = (int)(t / 128);
    _Float16 v = (_Float16)0.f;
    if (k < 8) {
        float f;
        if (r < 128) {                       // from y4, 2x2 spatial
            int ch = r & 31; int hw = r >> 5;
            f = y4[(((long)b * 32 + ch) * 8 + k) * 4 + hw];
        } else {                             // from x3, 4x4 spatial
            int r2 = r - 128; int ch = r2 & 31; int hw = r2 >> 5;
            f = x3[(((long)b * 32 + ch) * 8 + k) * 16 + hw];
        }
        v = (_Float16)f;
    }
    dst[i] = v;
}

// ---------------------------------------------------------------------------
// Elementwise / reduction kernels
// ---------------------------------------------------------------------------
__global__ void k_zerof(float* p, long n)
{
    long i = (long)blockIdx.x * blockDim.x + threadIdx.x;
    if (i < n) p[i] = 0.f;
}

__global__ void k_add(float* __restrict__ y, const float* __restrict__ s, long n)
{
    long i = (long)blockIdx.x * blockDim.x + threadIdx.x;
    if (i < n) y[i] += s[i];
}

__global__ void k_bn_stats(const float* __restrict__ x, float* mu, float* var,
                           int Bn, int C, long HW)
{
    int c = blockIdx.x;
    long per = (long)Bn * HW;
    float s = 0.f, s2 = 0.f;
    for (long i = threadIdx.x; i < per; i += blockDim.x) {
        long b = i / HW, p = i % HW;
        float v = x[((long)b * C + c) * HW + p];
        s += v; s2 += v * v;
    }
    __shared__ float sh[256], sh2[256];
    sh[threadIdx.x] = s; sh2[threadIdx.x] = s2;
    __syncthreads();
    for (int o = 128; o > 0; o >>= 1) {
        if ((int)threadIdx.x < o) {
            sh[threadIdx.x]  += sh[threadIdx.x + o];
            sh2[threadIdx.x] += sh2[threadIdx.x + o];
        }
        __syncthreads();
    }
    if (threadIdx.x == 0) {
        float m = sh[0] / (float)per;
        mu[c]  = m;
        var[c] = sh2[0] / (float)per - m * m;
    }
}

__global__ void k_bn_apply(float* __restrict__ x, const float* mu, const float* var,
                           const float* g, const float* b,
                           int C, long HW, long total, float eps)
{
    long i = (long)blockIdx.x * blockDim.x + threadIdx.x;
    if (i >= total) return;
    int c = (int)((i / HW) % C);
    x[i] = (x[i] - mu[c]) * rsqrtf(var[c] + eps) * g[c] + b[c];
}

// squash over capsule axis: y (Bn, ch, nj, HW), norm over nj
__global__ void k_squash(float* __restrict__ y, int Bn, int ch, int nj, long HW)
{
    long i = (long)blockIdx.x * blockDim.x + threadIdx.x;
    long total = (long)Bn * ch * HW;
    if (i >= total) return;
    long p = i % HW; long t = i / HW;
    int cc = (int)(t % ch); int b = (int)(t / ch);
    long base = ((long)(b * ch + cc) * nj) * HW + p;
    float n2 = 0.f;
    for (int j = 0; j < nj; ++j) { float v = y[base + (long)j * HW]; n2 += v * v; }
    float nn = sqrtf(n2);
    float sc = nn / (1.f + n2 + 1e-8f);
    for (int j = 0; j < nj; ++j) y[base + (long)j * HW] *= sc;
}

// ---------------------------------------------------------------------------
// 3D-conv capsule routing (per batch element): softmax over (hw,ch)=128 per
// ci, weighted sum over ci, elementwise squash, 3 iterations, entropy.
// yg: (128, 256=ch*8+nj, 32=ci, 2, 2); out: (128, 32, 8, 2, 2)
// ---------------------------------------------------------------------------
__global__ __launch_bounds__(256)
void k_route3d(const float* __restrict__ yg, float* __restrict__ out, float* entAcc)
{
    int b = blockIdx.x;
    const float* Yb = yg + (long)b * 256 * 32 * 4;
    __shared__ float Bmat[4][32][32];   // [hw][ch][ci]
    __shared__ float Km[4][32][32];     // routing coefficients
    __shared__ float S[4][8][32];       // squashed S_hat [hw][nj][ch]
    int tid = threadIdx.x;
    for (int i = tid; i < 4096; i += 256) ((float*)Bmat)[i] = 0.f;
    __syncthreads();
    float ent = 0.f;
    for (int it = 0; it < 3; ++it) {
        if (tid < 32) {                 // softmax over 128 entries per ci
            int ci = tid;
            float mx = -1e30f;
            for (int p = 0; p < 128; ++p) mx = fmaxf(mx, ((float*)Bmat)[p * 32 + ci]);
            float sum = 0.f;
            for (int p = 0; p < 128; ++p) {
                float e = __expf(((float*)Bmat)[p * 32 + ci] - mx);
                ((float*)Km)[p * 32 + ci] = e; sum += e;
            }
            float inv = 1.f / sum;
            for (int p = 0; p < 128; ++p) {
                float kk = ((float*)Km)[p * 32 + ci] * inv;
                ((float*)Km)[p * 32 + ci] = kk;
                if (it == 2) ent -= kk * __logf(kk + 1e-8f);
            }
        }
        __syncthreads();
        for (int i = tid; i < 1024; i += 256) {   // S + elementwise squash
            int ch = i & 31; int nj = (i >> 5) & 7; int hw = i >> 8;
            float s = 0.f;
            const float* xr = Yb + ((long)(ch * 8 + nj) * 32) * 4 + hw;
            for (int ci = 0; ci < 32; ++ci) s += Km[hw][ch][ci] * xr[ci * 4];
            float nn = fabsf(s);
            S[hw][nj][ch] = s * (nn / (1.f + nn * nn + 1e-8f));
        }
        __syncthreads();
        if (it < 2) {
            for (int i = tid; i < 4096; i += 256) {
                int ci = i & 31; int ch = (i >> 5) & 31; int hw = i >> 10;
                float d = 0.f;
                for (int nj = 0; nj < 8; ++nj)
                    d += S[hw][nj][ch] * Yb[((long)((ch * 8 + nj) * 32 + ci)) * 4 + hw];
                Bmat[hw][ch][ci] += d;
            }
            __syncthreads();
        }
    }
    if (tid < 32) atomicAdd(entAcc, ent * (1.f / (128.f * 32.f)));
    for (int i = tid; i < 1024; i += 256) {
        int ch = i & 31; int nj = (i >> 5) & 7; int hw = i >> 8;
        out[(((long)b * 32 + ch) * 8 + nj) * 4 + hw] = S[hw][nj][ch];
    }
}

// ---------------------------------------------------------------------------
// Digit-capsule dynamic routing (per batch element). uhat: (128,640,5,16).
// Writes dig (128,5,16), masked (128,16), accumulates entropy.
// ---------------------------------------------------------------------------
__global__ __launch_bounds__(256)
void k_caps_route(const float* __restrict__ uhat, const float* __restrict__ bias,
                  float* __restrict__ dig, float* __restrict__ masked,
                  float* entAcc)
{
    int b = blockIdx.x;
    const float* U = uhat + (long)b * 51200;
    __shared__ float bij[640][5];
    __shared__ float s[5][16];
    __shared__ float v[5][16];
    __shared__ float entsh;
    for (int i = threadIdx.x; i < 640 * 5; i += 256) ((float*)bij)[i] = 0.f;
    if (threadIdx.x == 0) entsh = 0.f;
    __syncthreads();
    for (int it = 0; it < 3; ++it) {
        if (threadIdx.x < 80) ((float*)s)[threadIdx.x] = bias[threadIdx.x];
        __syncthreads();
        float entLocal = 0.f;
        for (int r = threadIdx.x; r < 640; r += 256) {
            float mx = bij[r][0];
            for (int c = 1; c < 5; ++c) mx = fmaxf(mx, bij[r][c]);
            float e[5], sum = 0.f;
            for (int c = 0; c < 5; ++c) { e[c] = __expf(bij[r][c] - mx); sum += e[c]; }
            float inv = 1.f / sum;
            for (int c = 0; c < 5; ++c) {
                float cc = e[c] * inv;
                if (it == 2) entLocal -= cc * __logf(cc + 1e-8f);
                for (int o = 0; o < 16; ++o)
                    atomicAdd(&s[c][o], cc * U[r * 80 + c * 16 + o]);
            }
        }
        __syncthreads();
        if (threadIdx.x < 5) {          // squash each output capsule over o
            int c = threadIdx.x;
            float n2 = 0.f;
            for (int o = 0; o < 16; ++o) { float x = s[c][o]; n2 += x * x; }
            float nn = sqrtf(n2);
            float sc = nn / (1.f + n2 + 1e-8f);
            for (int o = 0; o < 16; ++o) v[c][o] = s[c][o] * sc;
        }
        __syncthreads();
        if (it < 2) {
            for (int r = threadIdx.x; r < 640; r += 256)
                for (int c = 0; c < 5; ++c) {
                    float d = 0.f;
                    for (int o = 0; o < 16; ++o) d += U[r * 80 + c * 16 + o] * v[c][o];
                    bij[r][c] += d;
                }
            __syncthreads();
        } else {
            atomicAdd(&entsh, entLocal);
            __syncthreads();
            if (threadIdx.x == 0) atomicAdd(entAcc, entsh * (1.f / (128.f * 640.f)));
            if (threadIdx.x < 80) dig[(long)b * 80 + threadIdx.x] = ((float*)v)[threadIdx.x];
            __syncthreads();
            if (threadIdx.x == 0) {     // argmax norm -> masked vector
                int best = 0; float bn = -1.f;
                for (int c = 0; c < 5; ++c) {
                    float n2 = 0.f;
                    for (int o = 0; o < 16; ++o) n2 += v[c][o] * v[c][o];
                    float nn = sqrtf(n2);
                    if (nn > bn) { bn = nn; best = c; }
                }
                for (int o = 0; o < 16; ++o) masked[b * 16 + o] = v[best][o];
            }
        }
    }
}

// ---------------------------------------------------------------------------
// Host-side orchestration
// ---------------------------------------------------------------------------
struct Ctx {
    hipStream_t s;
    float* stats;
    _Float16* w16;
    _Float16* col;
};

static void run_conv(Ctx& c, const float* x, int Bn, int Cin, int H, int W,
                     const float* w, const float* bias, int Cout, int stride,
                     float* y, int relu = 0)
{
    int Ho = (stride == 2) ? H / 2 : H;
    int Wo = Ho;
    int K = Cin * 9, Kp = (K + 31) & ~31;
    long n = (long)Cout * Kp;
    k_pack_w<<<(unsigned)((n + 255) / 256), 256, 0, c.s>>>(w, c.w16, Cout, K, Kp);
    long N = (long)Bn * Ho * Wo;
    n = N * Kp;
    k_im2col<<<(unsigned)((n + 255) / 256), 256, 0, c.s>>>(x, c.col, Bn, Cin, H, W,
                                                           Ho, Wo, stride, K, Kp);
    dim3 g((unsigned)((N + TILE_N - 1) / TILE_N), (unsigned)((Cout + TILE_M - 1) / TILE_M), 1);
    k_gemm_wmma<<<g, 256, 0, c.s>>>(c.w16, 0, c.col, 0, y, 0,
                                    Cout, (int)N, Kp, Ho * Wo, 0, bias, relu);
}

static void run_caps_conv(Ctx& c, const float* x, int Bn, int Cin, int H, int W,
                          const float* w, const float* b, int Cout, int stride,
                          int nj, float* y)
{
    run_conv(c, x, Bn, Cin, H, W, w, b, Cout, stride, y);
    int Ho = (stride == 2) ? H / 2 : H;
    long HW = (long)Ho * Ho;
    int ch = Cout / nj;
    long n = (long)Bn * ch * HW;
    k_squash<<<(unsigned)((n + 255) / 256), 256, 0, c.s>>>(y, Bn, ch, nj, HW);
}

static void run_bn(Ctx& c, float* x, int Bn, int C, long HW,
                   const float* g, const float* b, float eps)
{
    k_bn_stats<<<C, 256, 0, c.s>>>(x, c.stats, c.stats + C, Bn, C, HW);
    long tot = (long)Bn * C * HW;
    k_bn_apply<<<(unsigned)((tot + 255) / 256), 256, 0, c.s>>>(x, c.stats, c.stats + C,
                                                               g, b, C, HW, tot, eps);
}

static void run_convt(Ctx& c, const float* x, int Bn, int Cin, int Hp, int Wp,
                      int prepad, const float* w, const float* b, int Cout,
                      int stride, int Ho, float* y, int relu)
{
    int K = Cin * 9, Kp = (K + 31) & ~31;
    long n = (long)Cout * Kp;
    k_pack_wt<<<(unsigned)((n + 255) / 256), 256, 0, c.s>>>(w, c.w16, Cin, Cout, Kp);
    long N = (long)Bn * Ho * Ho;
    n = N * Kp;
    k_im2col_t<<<(unsigned)((n + 255) / 256), 256, 0, c.s>>>(x, c.col, Bn, Cin, Hp, Wp,
                                                             Ho, Ho, stride, prepad, K, Kp);
    dim3 g((unsigned)((N + TILE_N - 1) / TILE_N), (unsigned)((Cout + TILE_M - 1) / TILE_M), 1);
    k_gemm_wmma<<<g, 256, 0, c.s>>>(c.w16, 0, c.col, 0, y, 0,
                                    Cout, (int)N, Kp, Ho * Ho, 0, b, relu);
}

extern "C" void kernel_launch(void* const* d_in, const int* in_sizes, int n_in,
                              void* d_out, int out_size, void* d_ws, size_t ws_size,
                              hipStream_t stream)
{
    (void)in_sizes; (void)n_in; (void)out_size; (void)ws_size;
    auto F = [&](int i) -> const float* { return (const float*)d_in[i]; };
    // input order: x, conv_w, conv_b, bn_g, bn_b, b1(8), b2(8), b3(8), b4(6),
    //              c3d_w, c3d_b, Wcaps, caps_bias, dec_w, dec_b, dbn_g, dbn_b,
    //              d1_w, d1_b, d2_w, d2_b, d3_w, d3_b, d4_w, d4_b
    const float* x = F(0);

    // ---- workspace bump allocator (requires ~300 MB) ----
    char* ws = (char*)d_ws;
    long off = 0;
    auto alloc = [&](long bytes) -> char* {
        char* p = ws + off;
        off += (bytes + 255) & ~255L;
        return p;
    };
    float*    ACT0  = (float*)alloc(64L << 20);
    float*    ACT1  = (float*)alloc(64L << 20);
    float*    SKIP  = (float*)alloc(16L << 20);
    float*    YG    = (float*)alloc(18L << 20);
    _Float16* W16   = (_Float16*)alloc(8L << 20);
    _Float16* COL   = (_Float16*)alloc(80L << 20);
    float*    STATS = (float*)alloc(4096);
    float*    X3    = (float*)alloc(4L << 20);
    float*    UHAT  = (float*)alloc(28L << 20);
    _Float16* FE16  = (_Float16*)alloc(8L << 20);
    float*    MASK  = (float*)alloc(65536);
    _Float16* MSK16 = (_Float16*)alloc(65536);

    Ctx c{stream, STATS, W16, COL};

    float* dig = (float*)d_out;              // (128,5,16)
    float* dec = (float*)d_out + 10240;      // (128,1,32,32)
    float* ent = (float*)d_out + 141312;     // scalar
    k_zerof<<<1, 32, 0, stream>>>(ent, 1);

    // ---- stem: conv(1->128, s1) + batchnorm ----
    run_conv(c, x, 128, 1, 32, 32, F(1), F(2), 128, 1, ACT0);
    run_bn(c, ACT0, 128, 128, 1024, F(3), F(4), 1e-8f);

    float* cur = ACT0; float* alt = ACT1;
    auto swapb = [&]() { float* t = cur; cur = alt; alt = t; };

    auto run_block = [&](int base, int Cin, int Cout, int H, int nj) {
        run_caps_conv(c, cur, 128, Cin, H, H, F(base), F(base + 1), Cout, 2, nj, alt);
        swapb();
        int Ho = H / 2;
        long sz = (long)128 * Cout * Ho * Ho;
        hipMemcpyAsync(SKIP, cur, sz * 4, hipMemcpyDeviceToDevice, stream);
        for (int i = 1; i < 4; ++i) {
            run_caps_conv(c, cur, 128, Cout, Ho, Ho, F(base + 2 * i), F(base + 2 * i + 1),
                          Cout, 1, nj, alt);
            swapb();
        }
        k_add<<<(unsigned)((sz + 255) / 256), 256, 0, stream>>>(cur, SKIP, sz);
    };

    run_block(5, 128, 128, 32, 4);    // b1: 32x32 -> 16x16
    run_block(13, 128, 256, 16, 8);   // b2: 16x16 -> 8x8
    run_block(21, 256, 256, 8, 8);    // b3: 8x8  -> 4x4

    // save x3 = (128,256,4,4) capsule tensor for feats concat
    hipMemcpyAsync(X3, cur, 128L * 256 * 16 * 4, hipMemcpyDeviceToDevice, stream);

    // b4[0]: 256->256 s2 -> 2x2
    run_caps_conv(c, cur, 128, 256, 4, 4, F(29), F(30), 256, 2, 8, alt);
    swapb();
    long sz4 = 128L * 256 * 4;
    hipMemcpyAsync(SKIP, cur, sz4 * 4, hipMemcpyDeviceToDevice, stream);

    // ---- 3D caps conv + routing ----
    {
        k_pack_w<<<(unsigned)((256L * 32 + 255) / 256), 256, 0, stream>>>(F(35), W16, 256, 27, 32);
        long N = 128L * 32 * 4;                        // b * D * (h*w)
        long n = N * 32;
        k_im2col3d<<<(unsigned)((n + 255) / 256), 256, 0, stream>>>(cur, COL);
        dim3 g((unsigned)((N + TILE_N - 1) / TILE_N), 2, 1);
        k_gemm_wmma<<<g, 256, 0, stream>>>(W16, 0, COL, 0, YG, 0,
                                           256, (int)N, 32, 128, 0, F(36), 0);
        k_route3d<<<128, 256, 0, stream>>>(YG, alt, ent);
        swapb();   // cur = routed capsule tensor (128,32,8,2,2)
    }

    // b4[1], b4[2] + skip
    run_caps_conv(c, cur, 128, 256, 2, 2, F(31), F(32), 256, 1, 8, alt); swapb();
    run_caps_conv(c, cur, 128, 256, 2, 2, F(33), F(34), 256, 1, 8, alt); swapb();
    k_add<<<(unsigned)((sz4 + 255) / 256), 256, 0, stream>>>(cur, SKIP, sz4);

    // ---- digit capsules: batched WMMA einsum + routing ----
    {
        long n = 640L * 128 * 32;
        k_pack_feats<<<(unsigned)((n + 255) / 256), 256, 0, stream>>>(cur, X3, FE16);
        n = 51200L * 32;
        k_pack_w<<<(unsigned)((n + 255) / 256), 256, 0, stream>>>(F(37), W16, 51200, 8, 32);
        dim3 g(2, 1, 640);   // N=128 (2x64), M=80 (1x128 guarded), z=640 slices
        k_gemm_wmma<<<g, 256, 0, stream>>>(W16, 80L * 32, FE16, 128L * 32, UHAT, 80,
                                           80, 128, 32, 1, 51200, nullptr, 0);
        k_caps_route<<<128, 256, 0, stream>>>(UHAT, F(38), dig, MASK, ent);
    }

    // ---- decoder ----
    {
        k_pack_w<<<16, 256, 0, stream>>>(MASK, MSK16, 128, 16, 32);
        k_pack_w<<<128, 256, 0, stream>>>(F(39), W16, 1024, 16, 32);
        dim3 g(2, 8, 1);    // N=128, M=1024
        k_gemm_wmma<<<g, 256, 0, stream>>>(W16, 0, MSK16, 0, ACT0, 0,
                                           1024, 128, 32, 1, 0, F(40), 1);  // relu
        run_bn(c, ACT0, 128, 16, 64, F(41), F(42), 1e-5f);
        run_convt(c, ACT0, 128, 16, 8, 8, 0, F(43), F(44), 64, 1, 8,  ACT1, 0);
        run_convt(c, ACT1, 128, 64, 8, 8, 0, F(45), F(46), 32, 2, 15, ACT0, 0);
        run_convt(c, ACT0, 128, 32, 15, 15, 1, F(47), F(48), 16, 2, 31, ACT1, 0);
        run_convt(c, ACT1, 128, 16, 31, 31, 1, F(49), F(50), 1, 1, 32, dec, 1);
    }
}
